// StableLMAttention_76467597738057
// MI455X (gfx1250) — matrix operand order repaired and checked by reading, not compiled, not run on hardware
//
#include <hip/hip_runtime.h>
#include <hip/hip_bf16.h>
#include <cmath>
#include <cstdint>

#define DEV_INLINE __device__ __forceinline__

namespace {
constexpr int Bc = 2, Tc = 2048, Dc = 5120;
constexpr int HQc = 32, HKc = 8, HDc = 160, RDc = 40;
constexpr float EPSc = 1e-5f;
constexpr float ROPE_BASEc = 5000000.0f;
constexpr int BT  = Bc * Tc;     // 4096 rows
constexpr int NQ  = HQc * HDc;   // 5120
constexpr int NKV = HKc * HDc;   // 1280

typedef __bf16 bf16;
typedef __attribute__((ext_vector_type(16))) __bf16 v16bf;
typedef __attribute__((ext_vector_type(8)))  __bf16 v8bf;
typedef __attribute__((ext_vector_type(8)))  float  v8f;

DEV_INLINE v16bf cat8(v8bf lo, v8bf hi) {
  return __builtin_shufflevector(lo, hi, 0,1,2,3,4,5,6,7,8,9,10,11,12,13,14,15);
}
DEV_INLINE v8bf ld8(const bf16* p) { return *(const v8bf*)p; }

DEV_INLINE v8f vzero8() {
  v8f v;
  #pragma unroll
  for (int i = 0; i < 8; i++) v[i] = 0.f;
  return v;
}

// reductions across a 16-lane half (wave32 C-layout: lanes 0-15 hold rows 0-7,
// lanes 16-31 hold rows 8-15; XOR masks 1..8 stay within each half)
DEV_INLINE float half_max(float v) {
  v = fmaxf(v, __shfl_xor(v, 1));
  v = fmaxf(v, __shfl_xor(v, 2));
  v = fmaxf(v, __shfl_xor(v, 4));
  v = fmaxf(v, __shfl_xor(v, 8));
  return v;
}
DEV_INLINE float half_sum(float v) {
  v += __shfl_xor(v, 1);
  v += __shfl_xor(v, 2);
  v += __shfl_xor(v, 4);
  v += __shfl_xor(v, 8);
  return v;
}
} // namespace

// ---------------------------------------------------------------------------
// 1) fp32 (K x N) -> bf16 transposed (N x K), LDS-tiled for coalescing
// ---------------------------------------------------------------------------
__global__ __launch_bounds__(256) void k_transpose_bf16(
    const float* __restrict__ W, bf16* __restrict__ WT, int K, int N) {
  __shared__ float tile[32][33];
  const int n0 = blockIdx.x * 32, k0 = blockIdx.y * 32;
  const int tx = threadIdx.x, ty = threadIdx.y;  // (32, 8)
  #pragma unroll
  for (int i = 0; i < 32; i += 8)
    tile[ty + i][tx] = W[(size_t)(k0 + ty + i) * N + n0 + tx];
  __syncthreads();
  #pragma unroll
  for (int i = 0; i < 32; i += 8)
    WT[(size_t)(n0 + ty + i) * K + k0 + tx] = (bf16)tile[tx][ty + i];
}

// ---------------------------------------------------------------------------
// 2) RMSNorm(x)*w + b -> bf16 nx  (one 256-thread block per row of D=5120)
// ---------------------------------------------------------------------------
__global__ __launch_bounds__(256) void k_rmsnorm_x(
    const float* __restrict__ x, const float* __restrict__ w,
    const float* __restrict__ bias, bf16* __restrict__ nx) {
  const int row = blockIdx.x;
  const float* xr = x + (size_t)row * Dc;
  float ss = 0.f;
  for (int i = threadIdx.x; i < Dc; i += 256) { float v = xr[i]; ss += v * v; }
  __shared__ float red[256];
  red[threadIdx.x] = ss;
  __syncthreads();
  for (int s = 128; s > 0; s >>= 1) {
    if (threadIdx.x < s) red[threadIdx.x] += red[threadIdx.x + s];
    __syncthreads();
  }
  const float r = rsqrtf(red[0] / (float)Dc + EPSc);
  bf16* o = nx + (size_t)row * Dc;
  for (int i = threadIdx.x; i < Dc; i += 256)
    o[i] = (bf16)(xr[i] * r * w[i] + bias[i]);
}

// ---------------------------------------------------------------------------
// 3) bf16 WMMA GEMM: C(MxN,f32) = A(MxK,bf16 row-major) * Bt(NxK,bf16)^T
//    macro-tile 64x128, 4 waves, each wave 32x64 (2x4 WMMA tiles), K-step 32
// ---------------------------------------------------------------------------
__global__ __launch_bounds__(128) void k_gemm_bf16(
    const bf16* __restrict__ A, const bf16* __restrict__ Bt,
    float* __restrict__ C, int M, int N, int K) {
  (void)M;
  __shared__ alignas(16) bf16 As[64 * 32];
  __shared__ alignas(16) bf16 Bs[128 * 32];
  const int row0 = blockIdx.y * 64, col0 = blockIdx.x * 128;
  const int tid = threadIdx.x;
  const int wave = tid >> 5, lane = tid & 31;
  const int g = lane >> 4, lm = lane & 15;
  const int mbase = (wave & 1) * 32, nbase = (wave >> 1) * 64;

  v8f acc[2][4];
  #pragma unroll
  for (int i = 0; i < 2; i++)
    #pragma unroll
    for (int j = 0; j < 4; j++) acc[i][j] = vzero8();

  for (int k0 = 0; k0 < K; k0 += 32) {
    // cooperative 16B loads into LDS (A: 64x32, B: 128x32, row-major in tile)
    #pragma unroll
    for (int c = 0; c < 2; c++) {
      int e = (tid + c * 128) * 8, r = e >> 5, cc = e & 31;
      *(v8bf*)&As[r * 32 + cc] = ld8(&A[(size_t)(row0 + r) * K + k0 + cc]);
    }
    #pragma unroll
    for (int c = 0; c < 4; c++) {
      int e = (tid + c * 128) * 8, r = e >> 5, cc = e & 31;
      *(v8bf*)&Bs[r * 32 + cc] = ld8(&Bt[(size_t)(col0 + r) * K + k0 + cc]);
    }
    __syncthreads();

    v16bf a[2], b[4];
    #pragma unroll
    for (int i = 0; i < 2; i++) {  // A-layout: elems 0-7 K=g*8.., 8-15 K=16+g*8..
      const bf16* p = &As[(mbase + i * 16 + lm) * 32 + g * 8];
      a[i] = cat8(ld8(p), ld8(p + 16));
    }
    #pragma unroll
    for (int j = 0; j < 4; j++) {  // B-layout: col n=lm, elems 0-7 K=g*8.., 8-15 K=16+g*8..
      const bf16* p = &Bs[(nbase + j * 16 + lm) * 32 + g * 8];
      b[j] = cat8(ld8(p), ld8(p + 16));
    }
    #pragma unroll
    for (int i = 0; i < 2; i++)
      #pragma unroll
      for (int j = 0; j < 4; j++)
        acc[i][j] = __builtin_amdgcn_wmma_f32_16x16x32_bf16(
            false, a[i], false, b[j], (short)0, acc[i][j], false, false);
    __syncthreads();
  }
  #pragma unroll
  for (int i = 0; i < 2; i++)
    #pragma unroll
    for (int j = 0; j < 4; j++)
      #pragma unroll
      for (int r = 0; r < 8; r++)
        C[(size_t)(row0 + mbase + i * 16 + r + 8 * g) * N +
          col0 + nbase + j * 16 + lm] = acc[i][j][r];
}

// ---------------------------------------------------------------------------
// 4) per-head RMSNorm + partial RoPE, fp32 (bt, h, d) -> bf16 (b, h, t, d)
//    outscale folds 1/sqrt(HD) into Q (softmax-invariant)
// ---------------------------------------------------------------------------
__global__ __launch_bounds__(256) void k_headnorm_rope(
    const float* __restrict__ src, const float* __restrict__ nw,
    bf16* __restrict__ dst, int nheads, float outscale) {
  const int idx = blockIdx.x * 256 + threadIdx.x;
  if (idx >= BT * nheads) return;
  const int h = idx % nheads, bt = idx / nheads;
  const int b = bt / Tc, t = bt % Tc;
  const float* s = src + (size_t)bt * (nheads * HDc) + h * HDc;
  const float* w = nw + h * HDc;
  float ss = 0.f;
  for (int d = 0; d < HDc; d++) { float v = s[d]; ss += v * v; }
  const float r = rsqrtf(ss / (float)HDc + EPSc);
  bf16* o = dst + ((size_t)(b * nheads + h) * Tc + t) * HDc;
  const float lnb = __logf(ROPE_BASEc);
  #pragma unroll
  for (int p = 0; p < RDc / 2; p++) {   // rotate_every_two with emb = [f,f]
    const int d0 = 2 * p, d1 = d0 + 1;
    const float v0 = s[d0] * r * w[d0];
    const float v1 = s[d1] * r * w[d1];
    const float f0 = (float)t * __expf(-((float)(d0 % 20) / 20.f) * lnb);
    const float f1 = (float)t * __expf(-((float)(d1 % 20) / 20.f) * lnb);
    const float o0 = v0 * __cosf(f0) - v1 * __sinf(f0);
    const float o1 = v1 * __cosf(f1) + v0 * __sinf(f1);
    o[d0] = (bf16)(o0 * outscale);
    o[d1] = (bf16)(o1 * outscale);
  }
  for (int d = RDc; d < HDc; d++)
    o[d] = (bf16)(s[d] * r * w[d] * outscale);
}

// ---------------------------------------------------------------------------
// 5) V fp32 (bt, h, d) -> bf16 transposed (b, h, d, t) for P·V B-operands
// ---------------------------------------------------------------------------
__global__ __launch_bounds__(256) void k_vconv(
    const float* __restrict__ vf, bf16* __restrict__ vb) {
  const int idx = blockIdx.x * 256 + threadIdx.x;  // -> (b, h, d, t)
  const int t = idx % Tc;
  int rest = idx / Tc;
  const int d = rest % HDc; rest /= HDc;
  const int h = rest % HKc;
  const int b = rest / HKc;
  vb[idx] = (bf16)vf[(size_t)(b * Tc + t) * NKV + h * HDc + d];
}

// ---------------------------------------------------------------------------
// 6) causal flash attention, 1 wave per (b, h, 16-query tile)
//    scores: 5 WMMAs per 16x16 tile (HD=160); P via LDS -> A-layout; P·V: 10 WMMAs
// ---------------------------------------------------------------------------
__global__ __launch_bounds__(32) void k_attn(
    const bf16* __restrict__ Qb, const bf16* __restrict__ Kb,
    const bf16* __restrict__ Vb, bf16* __restrict__ Attn) {
  __shared__ alignas(16) bf16 Ps[16 * 32];
  const int qt = blockIdx.x, h = blockIdx.y, b = blockIdx.z;
  const int hk = h / (HQc / HKc);
  const int lane = threadIdx.x, g = lane >> 4, lm = lane & 15;
  const int q0 = qt * 16;
  const bf16* Q  = Qb + ((size_t)(b * HQc + h)  * Tc + q0) * HDc;
  const bf16* Kp = Kb +  (size_t)(b * HKc + hk) * Tc * HDc;
  const bf16* Vp = Vb +  (size_t)(b * HKc + hk) * HDc * Tc;

  v16bf qa[5];
  #pragma unroll
  for (int c = 0; c < 5; c++) {
    const bf16* p = &Q[lm * HDc + c * 32 + g * 8];
    qa[c] = cat8(ld8(p), ld8(p + 16));
  }
  v8f o[10];
  #pragma unroll
  for (int j = 0; j < 10; j++) o[j] = vzero8();
  float rmax[8], rsum[8];
  #pragma unroll
  for (int r = 0; r < 8; r++) { rmax[r] = -3.0e38f; rsum[r] = 0.f; }

  const int limit = q0 + 16;                 // keys 0..q0+15 can be live
  for (int kt = 0; kt < limit; kt += 32) {
    v8f s[2];
    #pragma unroll
    for (int hh = 0; hh < 2; hh++) {
      const int kbase = kt + hh * 16;
      if (kbase >= limit) {                  // fully masked half-tile
        #pragma unroll
        for (int r = 0; r < 8; r++) s[hh][r] = -3.0e38f;
        continue;
      }
      v8f sc = vzero8();
      #pragma unroll
      for (int c = 0; c < 5; c++) {
        const bf16* p = &Kp[(size_t)(kbase + lm) * HDc + c * 32 + g * 8];
        v16bf kop = cat8(ld8(p), ld8(p + 16));
        sc = __builtin_amdgcn_wmma_f32_16x16x32_bf16(
            false, qa[c], false, kop, (short)0, sc, false, false);
      }
      #pragma unroll
      for (int r = 0; r < 8; r++)            // causal mask: key > query
        if (kbase + lm > q0 + r + 8 * g) sc[r] = -3.0e38f;
      s[hh] = sc;
    }
    // online softmax (rows r+8*g per lane-half)
    float scl[8], p0[8], p1[8];
    #pragma unroll
    for (int r = 0; r < 8; r++) {
      const float mx = half_max(fmaxf(s[0][r], s[1][r]));
      const float nm = fmaxf(rmax[r], mx);
      scl[r] = __expf(rmax[r] - nm);
      p0[r]  = __expf(s[0][r] - nm);
      p1[r]  = __expf(s[1][r] - nm);
      rsum[r] = rsum[r] * scl[r] + half_sum(p0[r] + p1[r]);
      rmax[r] = nm;
    }
    #pragma unroll
    for (int j = 0; j < 10; j++)
      #pragma unroll
      for (int r = 0; r < 8; r++) o[j][r] *= scl[r];

    __syncthreads();                         // protect Ps from prior reads
    #pragma unroll
    for (int r = 0; r < 8; r++) {            // C-layout -> 16x32 bf16 tile
      const int m = r + 8 * g;
      Ps[m * 32 + lm]      = (bf16)p0[r];
      Ps[m * 32 + 16 + lm] = (bf16)p1[r];
    }
    __syncthreads();
    v16bf pa;
    {                                        // reload as A-layout
      const bf16* p = &Ps[lm * 32 + g * 8];
      pa = cat8(ld8(p), ld8(p + 16));
    }
    #pragma unroll
    for (int j = 0; j < 10; j++) {           // P(16x32) · V(32x16) per d-tile
      const bf16* p = &Vp[(size_t)(j * 16 + lm) * Tc + kt + g * 8];
      v16bf vop = cat8(ld8(p), ld8(p + 16));
      o[j] = __builtin_amdgcn_wmma_f32_16x16x32_bf16(
          false, pa, false, vop, (short)0, o[j], false, false);
    }
  }
  #pragma unroll
  for (int r = 0; r < 8; r++) rsum[r] = 1.0f / rsum[r];
  #pragma unroll
  for (int j = 0; j < 10; j++)
    #pragma unroll
    for (int r = 0; r < 8; r++) {
      const float v = o[j][r] * rsum[r];
      Attn[(size_t)(b * Tc + q0 + r + 8 * g) * Dc + h * HDc + j * 16 + lm] =
          (bf16)v;
    }
}

// ---------------------------------------------------------------------------
extern "C" void kernel_launch(void* const* d_in, const int* in_sizes, int n_in,
                              void* d_out, int out_size, void* d_ws,
                              size_t ws_size, hipStream_t stream) {
  (void)in_sizes; (void)n_in; (void)out_size; (void)ws_size;
  const float* x      = (const float*)d_in[0];
  const float* wq     = (const float*)d_in[1];
  const float* wk     = (const float*)d_in[2];
  const float* wv     = (const float*)d_in[3];
  const float* wo     = (const float*)d_in[4];
  const float* norm_w = (const float*)d_in[5];
  const float* norm_b = (const float*)d_in[6];
  const float* qn_w   = (const float*)d_in[7];
  const float* kn_w   = (const float*)d_in[8];

  char* ws = (char*)d_ws;
  size_t off = 0;
  auto take = [&](size_t bytes) {
    size_t o = off;
    off += (bytes + 255) & ~(size_t)255;
    return o;
  };
  bf16*  woT = (bf16*)(ws + take((size_t)Dc * Dc  * 2));
  bf16*  wqT = (bf16*)(ws + take((size_t)Dc * NQ  * 2));  // reused as attn buf
  bf16*  wkT = (bf16*)(ws + take((size_t)Dc * NKV * 2));
  bf16*  wvT = (bf16*)(ws + take((size_t)Dc * NKV * 2));
  bf16*  nx  = (bf16*)(ws + take((size_t)BT * Dc  * 2));
  float* qf  = (float*)(ws + take((size_t)BT * NQ  * 4));
  float* kf  = (float*)(ws + take((size_t)BT * NKV * 4));
  float* vf  = (float*)(ws + take((size_t)BT * NKV * 4));
  bf16*  qb  = (bf16*)(ws + take((size_t)BT * NQ  * 2));
  bf16*  kb  = (bf16*)(ws + take((size_t)BT * NKV * 2));
  bf16*  vb  = (bf16*)(ws + take((size_t)BT * NKV * 2));
  bf16*  attnb = wqT;  // wqT dead after the Q projection GEMM

  const dim3 tb(32, 8);
  k_transpose_bf16<<<dim3(NQ / 32,  Dc / 32), tb, 0, stream>>>(wq, wqT, Dc, NQ);
  k_transpose_bf16<<<dim3(NKV / 32, Dc / 32), tb, 0, stream>>>(wk, wkT, Dc, NKV);
  k_transpose_bf16<<<dim3(NKV / 32, Dc / 32), tb, 0, stream>>>(wv, wvT, Dc, NKV);
  k_transpose_bf16<<<dim3(Dc / 32,  Dc / 32), tb, 0, stream>>>(wo, woT, Dc, Dc);

  k_rmsnorm_x<<<BT, 256, 0, stream>>>(x, norm_w, norm_b, nx);

  k_gemm_bf16<<<dim3(NQ / 128,  BT / 64), 128, 0, stream>>>(nx, wqT, qf, BT, NQ,  Dc);
  k_gemm_bf16<<<dim3(NKV / 128, BT / 64), 128, 0, stream>>>(nx, wkT, kf, BT, NKV, Dc);
  k_gemm_bf16<<<dim3(NKV / 128, BT / 64), 128, 0, stream>>>(nx, wvT, vf, BT, NKV, Dc);

  k_headnorm_rope<<<(BT * HQc + 255) / 256, 256, 0, stream>>>(
      qf, qn_w, qb, HQc, 1.0f / sqrtf((float)HDc));
  k_headnorm_rope<<<(BT * HKc + 255) / 256, 256, 0, stream>>>(
      kf, kn_w, kb, HKc, 1.0f);
  k_vconv<<<(Bc * HKc * HDc * Tc) / 256, 256, 0, stream>>>(vf, vb);

  k_attn<<<dim3(Tc / 16, HQc, Bc), 32, 0, stream>>>(qb, kb, vb, attnb);

  k_gemm_bf16<<<dim3(Dc / 128, BT / 64), 128, 0, stream>>>(
      attnb, woT, (float*)d_out, BT, Dc, Dc);
}
